// MaskedFastDiffusionModel_3590592659668
// MI455X (gfx1250) — compile-verified, matrix operand order checked by reading, not run on hardware
//
#include <hip/hip_runtime.h>
#include <math.h>

// ---------------- types ----------------
typedef __attribute__((ext_vector_type(16))) __bf16 v16bf;
typedef __attribute__((ext_vector_type(8)))  float  v8f;
typedef __attribute__((ext_vector_type(4)))  unsigned int u32x4;
typedef __attribute__((ext_vector_type(8)))  int i32x8;
typedef __attribute__((ext_vector_type(4)))  int i32x4;

union FragBF { v16bf v; uint4 u[2]; };

#define BDIM 4
#define SDIM 2048
#define DDIM 2048
#define VDIM 32000
#define BS   (BDIM*SDIM)      // 8192
#define HDIM (DDIM/2)         // 1024
#define K1DIM (2*DDIM)        // 4096
#define NUM_TIMESTEPS 1000

#if defined(__has_builtin)
#  if __has_builtin(__builtin_amdgcn_tensor_load_to_lds)
#    define HAVE_TDM 1
#  else
#    define HAVE_TDM 0
#  endif
#else
#  define HAVE_TDM 0
#endif

// ---------------- helpers ----------------
__device__ __forceinline__ float bf2f(unsigned short h) {
    unsigned int u = ((unsigned int)h) << 16;
    float f; __builtin_memcpy(&f, &u, 4); return f;
}
__device__ __forceinline__ unsigned short f2bf(float f) {
    unsigned int u; __builtin_memcpy(&u, &f, 4);
    unsigned int lsb = (u >> 16) & 1u;
    u += 0x7fffu + lsb;
    return (unsigned short)(u >> 16);
}
__device__ __forceinline__ float gelu_exact(float x) {
    return 0.5f * x * (1.0f + erff(x * 0.70710678118654752f));
}
__device__ __forceinline__ float sigmoidf(float x) {
    return 1.0f / (1.0f + __expf(-x));
}

// ---------------- 1) fp32 -> bf16 convert ----------------
__global__ void f32_to_bf16_kernel(const float* __restrict__ in,
                                   unsigned short* __restrict__ out, int n) {
    int i = blockIdx.x * blockDim.x + threadIdx.x;
    if (i < n) out[i] = f2bf(in[i]);
}

// ---------------- 2) cumsum context + hidden bf16 ----------------
__global__ void ctx_kernel(const float* __restrict__ hidden,
                           unsigned short* __restrict__ hid_bf,
                           unsigned short* __restrict__ ctx_bf) {
    int idx = blockIdx.x * blockDim.x + threadIdx.x;   // 0 .. B*D-1
    if (idx >= BDIM * DDIM) return;
    int b = idx / DDIM, d = idx % DDIM;
    const float* base = hidden + (size_t)b * SDIM * DDIM + d;

    float total = 0.f;
    for (int s = 0; s < SDIM; ++s) total += base[(size_t)s * DDIM];

    float csum = 0.f;
    size_t o = (size_t)b * SDIM * DDIM + d;
    for (int s = 0; s < SDIM; ++s) {
        float x = base[(size_t)s * DDIM];
        float prev = csum;
        csum += x;
        float back = prev / fmaxf((float)s, 1.0f);
        float fwd  = (total - csum) / fmaxf((float)(SDIM - 1 - s), 1.0f);
        ctx_bf[o] = f2bf(0.5f * (fwd + back));
        hid_bf[o] = f2bf(x);
        o += DDIM;
    }
}

// ---------------- 3) bf16 WMMA GEMM (double-buffered pipeline) ----------------
#define TM 128
#define TN 128
#define TK 32
#define LDSTR 40   // LDS row stride in bf16 = 80B (TDM pad: 16DW interval + 4DW pad)

#if HAVE_TDM
__device__ __forceinline__ void tdm_load_A_tile(unsigned short* lds,
                                                const unsigned short* gsrc,
                                                int strideA) {
    unsigned long long ga = (unsigned long long)(size_t)gsrc;
    unsigned int lds_off  = (unsigned int)(size_t)lds;
    u32x4 g0;
    g0.x = 1u;                                   // count=1
    g0.y = lds_off;                              // lds_addr
    g0.z = (unsigned int)ga;                     // global_addr[31:0]
    g0.w = (unsigned int)((ga >> 32) & 0x01ffffffu) | (2u << 30); // addr[56:32], type=2
    i32x8 g1;
    // data_size=2B(enc1), pad_enable, pad_interval=16DW(enc3), pad_amount=4DW(enc3)
    g1[0] = (int)((1u << 16) | (1u << 20) | (3u << 22) | (3u << 25));
    g1[1] = (int)((TK & 0xffffu) << 16);          // tensor_dim0[15:0] @ 63:48
    g1[2] = (int)((TK >> 16) | ((TM & 0xffff) << 16));  // dim0 hi | dim1 lo
    g1[3] = (int)((TM >> 16) | (TK << 16));       // dim1 hi | tile_dim0=32
    g1[4] = (int)TM;                              // tile_dim1=128, tile_dim2=0
    g1[5] = (int)(unsigned)strideA;               // tensor_dim0_stride[31:0]
    g1[6] = 0;
    g1[7] = 0;
    i32x4 gz = {0, 0, 0, 0};
#if __clang_major__ >= 23
    i32x8 gz8 = {0, 0, 0, 0, 0, 0, 0, 0};
    __builtin_amdgcn_tensor_load_to_lds(g0, g1, gz, gz, gz8, 0);
#else
    __builtin_amdgcn_tensor_load_to_lds(g0, g1, gz, gz, 0);
#endif
}
#endif

__global__ __launch_bounds__(256) void gemm_bf16_kernel(
    const unsigned short* __restrict__ A0,
    const unsigned short* __restrict__ A1,
    int Ksplit, int K,
    const unsigned short* __restrict__ W,
    const float* __restrict__ bias,
    unsigned short* __restrict__ Out,
    int M, int N, int act)
{
    __shared__ unsigned short As[2][TM * LDSTR];  // [m][k], ping-pong
    __shared__ unsigned short Bs[2][TN * LDSTR];  // [n][k] transposed, ping-pong

    const int tid  = threadIdx.x;
    const int bm   = blockIdx.y, bn = blockIdx.x;
    const int wave = tid >> 5, lane = tid & 31;
    const int wm   = wave & 3;          // M offset wm*32
    const int wn   = wave >> 2;         // N offset wn*64
    const int l16  = lane & 15;
    const int khalf = (lane >> 4) * 8;
    const int K1 = K - Ksplit;

    v8f acc[2][4];
#pragma unroll
    for (int i = 0; i < 2; ++i)
#pragma unroll
        for (int j = 0; j < 4; ++j) {
            v8f z = {0.f, 0.f, 0.f, 0.f, 0.f, 0.f, 0.f, 0.f};
            acc[i][j] = z;
        }

    // uniform source selection for the [hid | ctx] concat
    auto srcA_of = [&](int k0, int& strideA) -> const unsigned short* {
        if (k0 < Ksplit) { strideA = Ksplit; return A0 + k0; }
        strideA = K1; return A1 + (k0 - Ksplit);
    };

    // W staging chunk decomposition (2 chunks per thread)
    const int c0 = tid, c1 = tid + 256;
    const int kr0 = c0 >> 4, cc0 = c0 & 15;
    const int kr1 = c1 >> 4, cc1 = c1 & 15;

    // ---------------- prologue: stage tile 0 into buffer 0 ----------------
    {
        int strideA; const unsigned short* sA = srcA_of(0, strideA);
#if HAVE_TDM
        if (tid < 32) tdm_load_A_tile(&As[0][0], sA + (size_t)(bm * TM) * strideA, strideA);
#else
        {
            uint4 av[2];
#pragma unroll
            for (int it = 0; it < 2; ++it) {
                int c = tid + it * 256; int r = c >> 2, ch = c & 3;
                av[it] = *(const uint4*)(sA + (size_t)(bm * TM + r) * strideA + ch * 8);
            }
#pragma unroll
            for (int it = 0; it < 2; ++it) {
                int c = tid + it * 256; int r = c >> 2, ch = c & 3;
                *(uint4*)(&As[0][0] + r * LDSTR + ch * 8) = av[it];
            }
        }
#endif
        uint4 wv0 = *(const uint4*)(W + (size_t)kr0 * N + bn * TN + cc0 * 8);
        uint4 wv1 = *(const uint4*)(W + (size_t)kr1 * N + bn * TN + cc1 * 8);
        unsigned short e[8];
        __builtin_memcpy(e, &wv0, 16);
#pragma unroll
        for (int t = 0; t < 8; ++t) Bs[0][(cc0 * 8 + t) * LDSTR + kr0] = e[t];
        __builtin_memcpy(e, &wv1, 16);
#pragma unroll
        for (int t = 0; t < 8; ++t) Bs[0][(cc1 * 8 + t) * LDSTR + kr1] = e[t];
#if HAVE_TDM
        if (tid < 32) __builtin_amdgcn_s_wait_tensorcnt(0);
#endif
        __syncthreads();
    }

    // ---------------- pipelined main loop ----------------
    int buf = 0;
    for (int k0 = 0; k0 < K; k0 += TK) {
        const int  kn = k0 + TK;
        const bool haveNext = kn < K;
        uint4 wv0, wv1;
#if !HAVE_TDM
        uint4 av[2];
#endif
        if (haveNext) {
            int strideA; const unsigned short* sA = srcA_of(kn, strideA);
#if HAVE_TDM
            if (tid < 32) tdm_load_A_tile(&As[buf ^ 1][0],
                                          sA + (size_t)(bm * TM) * strideA, strideA);
#else
#pragma unroll
            for (int it = 0; it < 2; ++it) {
                int c = tid + it * 256; int r = c >> 2, ch = c & 3;
                av[it] = *(const uint4*)(sA + (size_t)(bm * TM + r) * strideA + ch * 8);
            }
#endif
            wv0 = *(const uint4*)(W + (size_t)(kn + kr0) * N + bn * TN + cc0 * 8);
            wv1 = *(const uint4*)(W + (size_t)(kn + kr1) * N + bn * TN + cc1 * 8);
            if (kn + TK < K)
                __builtin_prefetch(W + (size_t)(kn + TK + (tid & 31)) * N + bn * TN, 0, 1);
        }

        // ---- compute current buffer: 12 ds_load_b128 + 8 WMMA ----
        const unsigned short* Ab = &As[buf][0];
        const unsigned short* Bb = &Bs[buf][0];
        FragBF a[2], b[4];
#pragma unroll
        for (int i = 0; i < 2; ++i) {
            int row = wm * 32 + i * 16 + l16;
            a[i].u[0] = *(const uint4*)(Ab + row * LDSTR + khalf);
            a[i].u[1] = *(const uint4*)(Ab + row * LDSTR + khalf + 16);
        }
#pragma unroll
        for (int j = 0; j < 4; ++j) {
            int col = wn * 64 + j * 16 + l16;
            b[j].u[0] = *(const uint4*)(Bb + col * LDSTR + khalf);
            b[j].u[1] = *(const uint4*)(Bb + col * LDSTR + khalf + 16);
        }
#pragma unroll
        for (int i = 0; i < 2; ++i)
#pragma unroll
            for (int j = 0; j < 4; ++j)
                acc[i][j] = __builtin_amdgcn_wmma_f32_16x16x32_bf16(
                    false, a[i].v, false, b[j].v, (short)0, acc[i][j], false, false);

        // ---- drain next-tile staging into buf^1 ----
        if (haveNext) {
#if !HAVE_TDM
#pragma unroll
            for (int it = 0; it < 2; ++it) {
                int c = tid + it * 256; int r = c >> 2, ch = c & 3;
                *(uint4*)(&As[buf ^ 1][0] + r * LDSTR + ch * 8) = av[it];
            }
#endif
            unsigned short e[8];
            __builtin_memcpy(e, &wv0, 16);
#pragma unroll
            for (int t = 0; t < 8; ++t) Bs[buf ^ 1][(cc0 * 8 + t) * LDSTR + kr0] = e[t];
            __builtin_memcpy(e, &wv1, 16);
#pragma unroll
            for (int t = 0; t < 8; ++t) Bs[buf ^ 1][(cc1 * 8 + t) * LDSTR + kr1] = e[t];
#if HAVE_TDM
            if (tid < 32) __builtin_amdgcn_s_wait_tensorcnt(0);
#endif
        }
        __syncthreads();
        buf ^= 1;
    }

    // ---------------- epilogue ----------------
    const int n_lane = lane & 15;
    const int mhi    = (lane >> 4) * 8;
#pragma unroll
    for (int i = 0; i < 2; ++i)
#pragma unroll
        for (int j = 0; j < 4; ++j) {
            int colg = bn * TN + wn * 64 + j * 16 + n_lane;
            float bv = bias[colg];
#pragma unroll
            for (int r = 0; r < 8; ++r) {
                int mg = bm * TM + wm * 32 + i * 16 + mhi + r;
                float v = acc[i][j][r] + bv;
                if (act) v = gelu_exact(v);
                Out[(size_t)mg * N + colg] = f2bf(v);
            }
        }
}

// ---------------- 4) wave-per-row GEMV + sigmoid ----------------
__global__ void gemv_sig_kernel(const unsigned short* __restrict__ X,
                                const float* __restrict__ w,
                                const float* __restrict__ bias,
                                float* __restrict__ out, int K) {
    int wave = threadIdx.x >> 5, lane = threadIdx.x & 31;
    int row = blockIdx.x * 8 + wave;
    if (row >= BS) return;
    const uint4* xp = (const uint4*)(X + (size_t)row * K);
    float acc = 0.f;
    for (int i = lane; i < (K >> 3); i += 32) {
        uint4 v = xp[i];
        unsigned short e[8]; __builtin_memcpy(e, &v, 16);
        const float* wp = w + i * 8;
#pragma unroll
        for (int t = 0; t < 8; ++t) acc += bf2f(e[t]) * wp[t];
    }
#pragma unroll
    for (int o = 16; o > 0; o >>= 1) acc += __shfl_down(acc, o, 32);
    if (lane == 0) out[row] = sigmoidf(acc + bias[0]);
}

// ---------------- 5) one-pass online softmax stats (NT loads) ----------------
__global__ __launch_bounds__(256) void softmax_stats_kernel(
    const float* __restrict__ logits,
    float* __restrict__ maxp, float* __restrict__ entc) {
    int row = blockIdx.x;
    int tid = threadIdx.x;
    const float* lp = logits + (size_t)row * VDIM;

    float m  = __builtin_nontemporal_load(lp + tid);
    float s1 = 1.f, s2 = 0.f;
    for (int i = tid + 256; i < VDIM; i += 256) {
        float x = __builtin_nontemporal_load(lp + i);
        if (x > m) {
            float e = __expf(m - x);
            s2 = (s2 + (m - x) * s1) * e;
            s1 = s1 * e + 1.f;
            m  = x;
        } else {
            float e = __expf(x - m);
            s1 += e;
            s2 += (x - m) * e;
        }
    }
    __shared__ float rm[256], r1[256], r2[256];
    rm[tid] = m; r1[tid] = s1; r2[tid] = s2;
    __syncthreads();
    for (int st = 128; st > 0; st >>= 1) {
        if (tid < st) {
            float mA = rm[tid],     s1A = r1[tid],     s2A = r2[tid];
            float mB = rm[tid+st],  s1B = r1[tid+st],  s2B = r2[tid+st];
            float mm = fmaxf(mA, mB);
            float eA = __expf(mA - mm), eB = __expf(mB - mm);
            rm[tid] = mm;
            r1[tid] = s1A * eA + s1B * eB;
            r2[tid] = (s2A + (mA - mm) * s1A) * eA + (s2B + (mB - mm) * s1B) * eB;
        }
        __syncthreads();
    }
    if (tid == 0) {
        float Z = r1[0];
        maxp[row] = 1.f / Z;
        float ent = __logf(Z) - r2[0] / Z;
        entc[row] = 1.f - ent / logf((float)VDIM);
    }
}

// ---------------- 6) adjacent-row cosine similarity ----------------
__global__ __launch_bounds__(256) void cossim_kernel(
    const float* __restrict__ hidden,
    float* __restrict__ norms, float* __restrict__ dots) {
    int row = blockIdx.x;
    int s   = row % SDIM;
    int tid = threadIdx.x;
    const float* r0 = hidden + (size_t)row * DDIM;
    bool hasNext = (s < SDIM - 1);
    float nq = 0.f, dt = 0.f;
    for (int i = tid; i < DDIM; i += 256) {
        float x = r0[i];
        nq += x * x;
        if (hasNext) dt += x * r0[i + DDIM];
    }
    __shared__ float rn[256], rd[256];
    rn[tid] = nq; rd[tid] = dt;
    __syncthreads();
    for (int st = 128; st > 0; st >>= 1) {
        if (tid < st) { rn[tid] += rn[tid+st]; rd[tid] += rd[tid+st]; }
        __syncthreads();
    }
    if (tid == 0) {
        norms[row] = sqrtf(rn[0]);
        dots[row]  = hasNext ? rd[0] : 0.f;
    }
}

// ---------------- 7) final combine -> d_out ----------------
__global__ void combine_kernel(const float* __restrict__ noise,
                               const float* __restrict__ maxp,
                               const float* __restrict__ entc,
                               const float* __restrict__ lconf,
                               const float* __restrict__ norms,
                               const float* __restrict__ dots,
                               const unsigned char* __restrict__ mask,
                               const int* __restrict__ gts,
                               float* __restrict__ out) {
    int i = blockIdx.x * blockDim.x + threadIdx.x;
    if (i >= BS) return;
    int s = i % SDIM;
    float mf = mask[i] ? 1.f : 0.f;
    float ts = (float)gts[0];

    float at = ts * (0.5f + noise[i]);
    at = at * (1.f + 0.3f * mf);
    at = fminf(fmaxf(at, 0.f), (float)(NUM_TIMESTEPS - 1));

    float ns = fmaxf(norms[i], 1e-12f);
    float prev = 0.f, next = 0.f;
    if (s > 0)        prev = dots[i - 1] / (fmaxf(norms[i - 1], 1e-12f) * ns);
    if (s < SDIM - 1) next = dots[i]     / (ns * fmaxf(norms[i + 1], 1e-12f));
    float cnt = (s == 0 || s == SDIM - 1) ? 1.f : 2.f;
    float boost = sigmoidf(2.f * (prev + next) / cnt);

    float conf = (0.4f * maxp[i] + 0.2f * entc[i] + 0.2f * lconf[i] + 0.2f * boost) * mf;
    out[i]      = at;
    out[BS + i] = conf;
}

// ---------------- launch ----------------
extern "C" void kernel_launch(void* const* d_in, const int* in_sizes, int n_in,
                              void* d_out, int out_size, void* d_ws, size_t ws_size,
                              hipStream_t stream) {
    const float*         hidden = (const float*)d_in[0];
    const float*         logits = (const float*)d_in[1];
    const unsigned char* mask   = (const unsigned char*)d_in[2];
    const int*           gts    = (const int*)d_in[3];
    const float* np_w1 = (const float*)d_in[4];
    const float* np_b1 = (const float*)d_in[5];
    const float* np_w2 = (const float*)d_in[6];
    const float* np_b2 = (const float*)d_in[7];
    const float* np_w3 = (const float*)d_in[8];
    const float* np_b3 = (const float*)d_in[9];
    const float* ch_w1 = (const float*)d_in[10];
    const float* ch_b1 = (const float*)d_in[11];
    const float* ch_w2 = (const float*)d_in[12];
    const float* ch_b2 = (const float*)d_in[13];
    float* out = (float*)d_out;

    char* ws = (char*)d_ws;
    size_t off = 0;
    auto carve = [&](size_t bytes) -> void* {
        void* p = ws + off;
        off = (off + bytes + 255) & ~(size_t)255;
        return p;
    };
    unsigned short* hid_bf = (unsigned short*)carve((size_t)BS * DDIM * 2);
    unsigned short* ctx_bf = (unsigned short*)carve((size_t)BS * DDIM * 2);
    unsigned short* h1_bf  = (unsigned short*)carve((size_t)BS * DDIM * 2);
    unsigned short* h2_bf  = (unsigned short*)carve((size_t)BS * HDIM * 2);
    unsigned short* c_bf   = (unsigned short*)carve((size_t)BS * HDIM * 2);
    unsigned short* w1_bf  = (unsigned short*)carve((size_t)K1DIM * DDIM * 2);
    unsigned short* w2_bf  = (unsigned short*)carve((size_t)DDIM * HDIM * 2);
    unsigned short* cw1_bf = (unsigned short*)carve((size_t)DDIM * HDIM * 2);
    float* noise  = (float*)carve((size_t)BS * 4);
    float* maxp   = (float*)carve((size_t)BS * 4);
    float* entc   = (float*)carve((size_t)BS * 4);
    float* lconf  = (float*)carve((size_t)BS * 4);
    float* norms  = (float*)carve((size_t)BS * 4);
    float* dots   = (float*)carve((size_t)BS * 4);

    {
        int n1 = K1DIM * DDIM, n2 = DDIM * HDIM;
        f32_to_bf16_kernel<<<(n1 + 255) / 256, 256, 0, stream>>>(np_w1, w1_bf, n1);
        f32_to_bf16_kernel<<<(n2 + 255) / 256, 256, 0, stream>>>(np_w2, w2_bf, n2);
        f32_to_bf16_kernel<<<(n2 + 255) / 256, 256, 0, stream>>>(ch_w1, cw1_bf, n2);
    }
    ctx_kernel<<<(BDIM * DDIM) / 256, 256, 0, stream>>>(hidden, hid_bf, ctx_bf);

    gemm_bf16_kernel<<<dim3(DDIM / TN, BS / TM), 256, 0, stream>>>(
        hid_bf, ctx_bf, DDIM, K1DIM, w1_bf, np_b1, h1_bf, BS, DDIM, 1);
    gemm_bf16_kernel<<<dim3(HDIM / TN, BS / TM), 256, 0, stream>>>(
        h1_bf, nullptr, DDIM, DDIM, w2_bf, np_b2, h2_bf, BS, HDIM, 1);
    gemv_sig_kernel<<<BS / 8, 256, 0, stream>>>(h2_bf, np_w3, np_b3, noise, HDIM);

    gemm_bf16_kernel<<<dim3(HDIM / TN, BS / TM), 256, 0, stream>>>(
        hid_bf, nullptr, DDIM, DDIM, cw1_bf, ch_b1, c_bf, BS, HDIM, 1);
    gemv_sig_kernel<<<BS / 8, 256, 0, stream>>>(c_bf, ch_w2, ch_b2, lconf, HDIM);

    softmax_stats_kernel<<<BS, 256, 0, stream>>>(logits, maxp, entc);
    cossim_kernel<<<BS, 256, 0, stream>>>(hidden, norms, dots);
    combine_kernel<<<BS / 256, 256, 0, stream>>>(noise, maxp, entc, lconf,
                                                 norms, dots, mask, gts, out);
}
